// Encoder_88691074663154
// MI455X (gfx1250) — compile-verified, matrix operand order checked by reading, not run on hardware
//
#include <hip/hip_runtime.h>
#include <math.h>

// Transformer encoder layer for MI455X (gfx1250, wave32, WMMA).
// All matmuls run on v_wmma_f32_16x16x32_bf16 (bf16 in, f32 accumulate).

#define LSEQ 2048
#define BB   4
#define EE   512
#define HH   8
#define HDIM 64
#define HID  2048
#define MTOK (LSEQ*BB)   // 8192 tokens
#define E3   (3*EE)      // 1536

typedef __attribute__((ext_vector_type(16))) __bf16 v16bf;
typedef __attribute__((ext_vector_type(8)))  __bf16 v8bf;
typedef __attribute__((ext_vector_type(8)))  float  v8f;

static __device__ inline v8f vzero8() {
  v8f r;
  #pragma unroll
  for (int i = 0; i < 8; ++i) r[i] = 0.0f;
  return r;
}

static __device__ inline v16bf mk16(v8bf lo, v8bf hi) {
  v16bf r;
  #pragma unroll
  for (int i = 0; i < 8; ++i) { r[i] = lo[i]; r[i + 8] = hi[i]; }
  return r;
}

// Fragment pattern (A-rows and B-rows, 16-bit WMMA operands): lane holds
// row/col (lane&15); halves are K = koff+[0..7] and koff+16+[0..7],
// koff = (lane<16) ? 0 : 8  ->  two 16-byte loads from one row.
static __device__ inline v16bf load_frag(const __bf16* rowp) {
  return mk16(*reinterpret_cast<const v8bf*>(rowp),
              *reinterpret_cast<const v8bf*>(rowp + 16));
}

#define WMMA_BF16(a, b, c) \
  __builtin_amdgcn_wmma_f32_16x16x32_bf16(false, (a), false, (b), (short)0, (c), false, false)

// ---- half-wave (16-lane) butterfly reductions: DPP quad_perm for xor1/xor2,
// ---- ds_swizzle (group-of-32, xor mode) for xor4/xor8. No bpermute traffic.
#define DPPF(x, ctrl) \
  __builtin_bit_cast(float, __builtin_amdgcn_update_dpp(0, __builtin_bit_cast(int, (x)), (ctrl), 0xf, 0xf, true))
#define SWZF(x, imm) \
  __builtin_bit_cast(float, __builtin_amdgcn_ds_swizzle(__builtin_bit_cast(int, (x)), (imm)))

static __device__ inline float redmax16(float x) {
  x = fmaxf(x, DPPF(x, 0xB1));      // quad_perm [1,0,3,2]  (xor 1)
  x = fmaxf(x, DPPF(x, 0x4E));      // quad_perm [2,3,0,1]  (xor 2)
  x = fmaxf(x, SWZF(x, 0x101f));    // xor 4
  x = fmaxf(x, SWZF(x, 0x201f));    // xor 8
  return x;
}
static __device__ inline float redsum16(float x) {
  x += DPPF(x, 0xB1);
  x += DPPF(x, 0x4E);
  x += SWZF(x, 0x101f);
  x += SWZF(x, 0x201f);
  return x;
}
static __device__ inline float redsum32(float x) {  // full-wave sum (layernorm)
  x += DPPF(x, 0xB1);
  x += DPPF(x, 0x4E);
  x += SWZF(x, 0x101f);
  x += SWZF(x, 0x201f);
  x += SWZF(x, 0x401f);             // xor 16
  return x;
}

// ---------------------------------------------------------------- utilities
__global__ __launch_bounds__(256)
void cvt_bf16(const float* __restrict__ in, __bf16* __restrict__ out, int n) {
  int i = blockIdx.x * 256 + threadIdx.x;
  if (i < n) out[i] = (__bf16)in[i];
}

// x + positional encoding (pe varies along B per reference); writes f32
// residual stream and bf16 GEMM operand.
__global__ __launch_bounds__(256)
void pe_add(const float* __restrict__ x, float* __restrict__ x0,
            __bf16* __restrict__ xb) {
  int i = blockIdx.x * 256 + threadIdx.x;       // over MTOK*EE
  int e = i & (EE - 1);
  int b = (i >> 9) & (BB - 1);
  float ef  = (float)((e & 1) ? (e + 1) : e);
  float p   = __expf(ef * (-9.210340371976184f / (float)EE));   // 10000^(-ef/E)
  float ang = (float)(b + 1) * p;
  float pe  = (e & 1) ? cosf(ang) : sinf(ang);
  float v   = x[i] + pe;
  x0[i] = v;
  xb[i] = (__bf16)v;
}

// qkv (M,1536) bf16 -> vT (B*H*HD, L) bf16 so P@V B-fragments load contiguously.
__global__ __launch_bounds__(256)
void transpose_v(const __bf16* __restrict__ qkv, __bf16* __restrict__ vT) {
  int i = blockIdx.x * 256 + threadIdx.x;       // over MTOK*EE
  int e = i & (EE - 1);
  int t = i >> 9;                               // token m = l*B + b
  int b = t & (BB - 1);
  int l = t >> 2;
  vT[((size_t)b * EE + e) * LSEQ + l] = qkv[(size_t)t * E3 + 2 * EE + e];
}

// ------------------------------------------------------------------- GEMM
// C(M,N) = A(M,K)bf16 @ W(N,K)bf16^T, wave tile 32x64 (2x4 WMMA accum),
// block = 8 waves -> 64x256 tile.  K%64==0, double-buffered fragments.
// EPI: 0 = +bias store bf16 | 1 = +bias +resid store f32 | 2 = +bias GELU bf16
template <int EPI>
__global__ __launch_bounds__(256) __attribute__((amdgpu_waves_per_eu(1)))
void gemm_bf16(const __bf16* __restrict__ A, const __bf16* __restrict__ W,
               const float* __restrict__ bias, const float* __restrict__ resid,
               float* __restrict__ outF, __bf16* __restrict__ outB,
               int N, int K) {
  const int lane = threadIdx.x & 31;
  const int w    = threadIdx.x >> 5;
  const int m0   = blockIdx.y * 64  + (w >> 2) * 32;
  const int n0   = blockIdx.x * 256 + (w & 3) * 64;
  const int ridx = lane & 15;
  const int koff = (lane < 16) ? 0 : 8;

  // marching base pointers; per-iteration advance is a constant
  const __bf16* Ap0 = A + (size_t)(m0 + ridx) * K + koff;
  const __bf16* Ap1 = A + (size_t)(m0 + 16 + ridx) * K + koff;
  const __bf16* Wp0 = W + (size_t)(n0 + ridx) * K + koff;       // +16j*K immediates
  const size_t  WS  = (size_t)16 * K;                           // row-group stride

  v8f acc[2][4];
  #pragma unroll
  for (int i = 0; i < 2; ++i)
    #pragma unroll
    for (int j = 0; j < 4; ++j) acc[i][j] = vzero8();

  v16bf a0[2], b0[4], a1[2], b1[4];
  a0[0] = load_frag(Ap0); a0[1] = load_frag(Ap1);
  #pragma unroll
  for (int j = 0; j < 4; ++j) b0[j] = load_frag(Wp0 + (size_t)j * WS);

  for (int k0 = 0; k0 < K; k0 += 64) {
    // loads for second half-slab issue before first slab's WMMAs retire
    a1[0] = load_frag(Ap0 + 32); a1[1] = load_frag(Ap1 + 32);
    #pragma unroll
    for (int j = 0; j < 4; ++j) b1[j] = load_frag(Wp0 + (size_t)j * WS + 32);
    #pragma unroll
    for (int i = 0; i < 2; ++i)
      #pragma unroll
      for (int j = 0; j < 4; ++j)
        acc[i][j] = WMMA_BF16(a0[i], b0[j], acc[i][j]);
    if (k0 + 64 < K) {
      a0[0] = load_frag(Ap0 + 64); a0[1] = load_frag(Ap1 + 64);
      #pragma unroll
      for (int j = 0; j < 4; ++j) b0[j] = load_frag(Wp0 + (size_t)j * WS + 64);
    }
    #pragma unroll
    for (int i = 0; i < 2; ++i)
      #pragma unroll
      for (int j = 0; j < 4; ++j)
        acc[i][j] = WMMA_BF16(a1[i], b1[j], acc[i][j]);
    Ap0 += 64; Ap1 += 64; Wp0 += 64;
  }

  // C layout: lane holds col (lane&15); VGPR r -> row r + ((lane>=16)?8:0)
  const int ncol  = lane & 15;
  const int rbase = (lane >> 4) << 3;
  #pragma unroll
  for (int j = 0; j < 4; ++j) {
    int   n  = n0 + 16 * j + ncol;
    float bv = bias[n];
    #pragma unroll
    for (int i = 0; i < 2; ++i) {
      #pragma unroll
      for (int r = 0; r < 8; ++r) {
        int    m = m0 + 16 * i + rbase + r;
        size_t o = (size_t)m * N + n;
        float  v = acc[i][j][r] + bv;
        if (EPI == 1) { v += resid[o]; outF[o] = v; }
        else if (EPI == 2) {
          v = 0.5f * v * (1.0f + erff(v * 0.7071067811865475f));
          outB[o] = (__bf16)v;
        } else {
          outB[o] = (__bf16)v;
        }
      }
    }
  }
}

// -------------------------------------------------------------- attention
// One wave per (16 queries, b, h).  Streams 64 keys/step, online softmax.
// V-fragment loads split around the softmax / pa0-WMMAs to cap VGPR pressure.
__global__ __launch_bounds__(32) __attribute__((amdgpu_waves_per_eu(1)))
void attention(const __bf16* __restrict__ qkv, const __bf16* __restrict__ vT,
               __bf16* __restrict__ aout) {
  __shared__ __align__(16) __bf16 pbuf[16 * 72];   // 16x64 P tile, padded rows
  const int lane  = threadIdx.x;                   // block of 32 (one wave)
  const int l0    = blockIdx.x * 16;
  const int bh    = blockIdx.y;
  const int b     = bh >> 3;
  const int h     = bh & 7;
  const int ridx  = lane & 15;
  const int koff  = (lane < 16) ? 0 : 8;
  const int ncol  = ridx;
  const int rbase = (lane >> 4) << 3;

  v16bf qa[2];                                     // Q 16x64 A-fragments
  {
    const __bf16* qr = qkv + ((size_t)(l0 + ridx) * BB + b) * E3 + h * HDIM + koff;
    qa[0] = mk16(*(const v8bf*)qr,        *(const v8bf*)(qr + 16));
    qa[1] = mk16(*(const v8bf*)(qr + 32), *(const v8bf*)(qr + 48));
  }
  const size_t  KROW = (size_t)BB * E3;            // elems between key rows
  const __bf16* kptr = qkv + (size_t)ridx * KROW + (size_t)b * E3 + EE + h * HDIM + koff;
  const __bf16* vptr = vT + ((size_t)bh * HDIM + ridx) * LSEQ + koff;

  v8f o[4];
  #pragma unroll
  for (int j = 0; j < 4; ++j) o[j] = vzero8();
  v8f mcur, lsum = vzero8();
  #pragma unroll
  for (int r = 0; r < 8; ++r) mcur[r] = -3.0e38f;

  for (int kk = 0; kk < LSEQ; kk += 64) {
    v8f s[4];
    {
      v16bf kb[4][2];                              // transient: K fragments
      #pragma unroll
      for (int t = 0; t < 4; ++t) {
        const __bf16* kr = kptr + (size_t)(16 * t) * KROW;
        kb[t][0] = mk16(*(const v8bf*)kr,        *(const v8bf*)(kr + 16));
        kb[t][1] = mk16(*(const v8bf*)(kr + 32), *(const v8bf*)(kr + 48));
      }
      #pragma unroll
      for (int t = 0; t < 4; ++t) {                // scores, 2 WMMA per key tile
        v8f sv = vzero8();
        sv = WMMA_BF16(qa[0], kb[t][0], sv);
        sv = WMMA_BF16(qa[1], kb[t][1], sv);
        #pragma unroll
        for (int r = 0; r < 8; ++r) s[t][r] = sv[r] * 0.125f;  // 1/sqrt(64)
      }
    }
    // first-half V fragments: loads complete during the softmax VALU work
    v16bf vb0[4];
    #pragma unroll
    for (int j = 0; j < 4; ++j) vb0[j] = load_frag(vptr + (size_t)(16 * j) * LSEQ);

    // online softmax: one half-wave reduction per 64 keys; p computed in
    // place over s to keep live VGPRs down.
    v8f scale;
    #pragma unroll
    for (int r = 0; r < 8; ++r) {
      float sm = fmaxf(fmaxf(s[0][r], s[1][r]), fmaxf(s[2][r], s[3][r]));
      sm = redmax16(sm);
      float nm = fmaxf(mcur[r], sm);
      scale[r] = __expf(mcur[r] - nm);
      mcur[r]  = nm;
      float ps = 0.0f;
      #pragma unroll
      for (int t = 0; t < 4; ++t) { s[t][r] = __expf(s[t][r] - nm); ps += s[t][r]; }
      lsum[r] = lsum[r] * scale[r] + ps;
    }
    #pragma unroll
    for (int j = 0; j < 4; ++j)
      #pragma unroll
      for (int r = 0; r < 8; ++r) o[j][r] *= scale[r];

    __syncthreads();                               // prior pbuf reads done
    #pragma unroll
    for (int t = 0; t < 4; ++t)
      #pragma unroll
      for (int r = 0; r < 8; ++r)
        pbuf[(rbase + r) * 72 + 16 * t + ncol] = (__bf16)s[t][r];
    __syncthreads();
    v16bf pa0 = load_frag(pbuf + ridx * 72 + koff);        // keys kk..kk+31
    v16bf pa1 = load_frag(pbuf + ridx * 72 + 32 + koff);   // keys kk+32..kk+63

    #pragma unroll
    for (int j = 0; j < 4; ++j) o[j] = WMMA_BF16(pa0, vb0[j], o[j]);
    // second-half V fragments: latency hides behind the pa0 WMMAs
    v16bf vb1[4];
    #pragma unroll
    for (int j = 0; j < 4; ++j) vb1[j] = load_frag(vptr + (size_t)(16 * j) * LSEQ + 32);
    #pragma unroll
    for (int j = 0; j < 4; ++j) o[j] = WMMA_BF16(pa1, vb1[j], o[j]);

    kptr += (size_t)64 * KROW;
    vptr += 64;
  }

  v8f inv;
  #pragma unroll
  for (int r = 0; r < 8; ++r) inv[r] = 1.0f / redsum16(lsum[r]);

  #pragma unroll
  for (int j = 0; j < 4; ++j)
    #pragma unroll
    for (int r = 0; r < 8; ++r) {
      int l = l0 + rbase + r;
      aout[((size_t)l * BB + b) * EE + h * HDIM + 16 * j + ncol] =
          (__bf16)(o[j][r] * inv[r]);
    }
}

// ------------------------------------------------------------- layernorm
// One wave per row of 512; writes f32 (residual / output) and optional bf16.
__global__ __launch_bounds__(256)
void layernorm(const float* __restrict__ in, const float* __restrict__ g,
               const float* __restrict__ bt, float* __restrict__ outF,
               __bf16* __restrict__ outB) {
  const int lane = threadIdx.x & 31;
  const int row  = blockIdx.x * 8 + (threadIdx.x >> 5);
  const float* rp = in + (size_t)row * EE;
  float v[16], s = 0.0f, s2 = 0.0f;
  #pragma unroll
  for (int i = 0; i < 16; ++i) {
    float t = rp[lane + 32 * i];
    v[i] = t; s += t; s2 += t * t;
  }
  s  = redsum32(s);
  s2 = redsum32(s2);
  float mu  = s * (1.0f / EE);
  float var = s2 * (1.0f / EE) - mu * mu;
  float inv = rsqrtf(var + 1e-5f);
  #pragma unroll
  for (int i = 0; i < 16; ++i) {
    int   e = lane + 32 * i;
    float t = (v[i] - mu) * inv * g[e] + bt[e];
    outF[(size_t)row * EE + e] = t;
    if (outB) outB[(size_t)row * EE + e] = (__bf16)t;
  }
}

// ---------------------------------------------------------------- launch
extern "C" void kernel_launch(void* const* d_in, const int* in_sizes, int n_in,
                              void* d_out, int out_size, void* d_ws, size_t ws_size,
                              hipStream_t stream) {
  const float* x     = (const float*)d_in[0];
  const float* in_w  = (const float*)d_in[1];
  const float* in_b  = (const float*)d_in[2];
  const float* out_w = (const float*)d_in[3];
  const float* out_b = (const float*)d_in[4];
  const float* w1    = (const float*)d_in[5];
  const float* b1    = (const float*)d_in[6];
  const float* w2    = (const float*)d_in[7];
  const float* b2    = (const float*)d_in[8];
  const float* ln_g  = (const float*)d_in[9];
  const float* ln_b  = (const float*)d_in[10];
  (void)in_sizes; (void)n_in; (void)out_size; (void)ws_size;

  char* ws = (char*)d_ws;
  size_t off = 0;
  auto alloc = [&](size_t bytes) {
    char* p = ws + off;
    off += (bytes + 255) & ~(size_t)255;
    return p;
  };
  float*  x0f   = (float*) alloc((size_t)MTOK * EE * 4);   // residual stream
  __bf16* wqkvb = (__bf16*)alloc((size_t)E3 * EE * 2);
  __bf16* woutb = (__bf16*)alloc((size_t)EE * EE * 2);
  __bf16* w1b   = (__bf16*)alloc((size_t)HID * EE * 2);
  __bf16* w2b   = (__bf16*)alloc((size_t)EE * HID * 2);
  __bf16* xb    = (__bf16*)alloc((size_t)MTOK * EE * 2);   // later reused as aout
  __bf16* qkvb  = (__bf16*)alloc((size_t)MTOK * E3 * 2);   // front reused as x1b
  __bf16* vT    = (__bf16*)alloc((size_t)MTOK * EE * 2);
  float*  y1f   = (float*) alloc((size_t)MTOK * EE * 4);   // reused as y2f
  float*  x1f   = (float*) alloc((size_t)MTOK * EE * 4);
  __bf16* hb    = (__bf16*)alloc((size_t)MTOK * HID * 2);
  __bf16* aout  = xb;      // xb dead after QKV GEMM
  __bf16* x1b   = qkvb;    // qkvb dead after attention
  float*  y2f   = y1f;     // y1f dead after LN1

  cvt_bf16<<<(E3 * EE + 255) / 256, 256, 0, stream>>>(in_w, wqkvb, E3 * EE);
  cvt_bf16<<<(EE * EE + 255) / 256, 256, 0, stream>>>(out_w, woutb, EE * EE);
  cvt_bf16<<<(HID * EE + 255) / 256, 256, 0, stream>>>(w1, w1b, HID * EE);
  cvt_bf16<<<(EE * HID + 255) / 256, 256, 0, stream>>>(w2, w2b, EE * HID);

  pe_add<<<(MTOK * EE) / 256, 256, 0, stream>>>(x, x0f, xb);

  gemm_bf16<0><<<dim3(E3 / 256, MTOK / 64), 256, 0, stream>>>(
      xb, wqkvb, in_b, nullptr, nullptr, qkvb, E3, EE);

  transpose_v<<<(MTOK * EE) / 256, 256, 0, stream>>>(qkvb, vT);

  attention<<<dim3(LSEQ / 16, BB * HH), 32, 0, stream>>>(qkvb, vT, aout);

  gemm_bf16<1><<<dim3(EE / 256, MTOK / 64), 256, 0, stream>>>(
      aout, woutb, out_b, x0f, y1f, nullptr, EE, EE);

  layernorm<<<MTOK / 8, 256, 0, stream>>>(y1f, ln_g, ln_b, x1f, x1b);

  gemm_bf16<2><<<dim3(HID / 256, MTOK / 64), 256, 0, stream>>>(
      x1b, w1b, b1, nullptr, nullptr, hb, HID, EE);

  gemm_bf16<1><<<dim3(EE / 256, MTOK / 64), 256, 0, stream>>>(
      hb, w2b, b2, x1f, y2f, nullptr, EE, HID);

  layernorm<<<MTOK / 8, 256, 0, stream>>>(y2f, ln_g, ln_b, (float*)d_out, nullptr);
}
